// Attention_1047972020922
// MI455X (gfx1250) — matrix operand order repaired and checked by reading, not compiled, not run on hardware
//
#include <hip/hip_runtime.h>
#include <hip/hip_bf16.h>

#define C_DIM 512
#define HW 1024
#define B_DIM 32
#define GROUPS 8
#define AROWS 32

typedef __attribute__((ext_vector_type(16))) _Float16 v16h;
typedef __attribute__((ext_vector_type(8)))  _Float16 v8h;
typedef __attribute__((ext_vector_type(8)))  float    v8f;

// ---------------- WMMA fragment helpers (CDNA5 16x16x32 f16, wave32) ----------
// A fragment: row-major A[M][K] tile at (m0,k0). Lane lo holds row m0+lo.
// Half e -> K = k0 + hi*16 + e  (16 contiguous f16)
__device__ __forceinline__ v16h load_a_frag(const _Float16* __restrict__ A, int lda,
                                            int m0, int k0, int lane) {
    const int lo = lane & 15, hi = lane >> 4;
    const _Float16* p = A + (size_t)(m0 + lo) * lda + k0 + hi * 16;
    v8h c0 = *(const v8h*)(p);
    v8h c1 = *(const v8h*)(p + 8);
    v16h a;
#pragma unroll
    for (int i = 0; i < 8; ++i) { a[i] = c0[i]; a[8 + i] = c1[i]; }
    return a;
}

// B fragment from B-transposed (row-major Bt[N][K]). Lane lo holds col n0+lo.
// Half e -> K = k0 + hi*16 + e  (16 contiguous f16)
__device__ __forceinline__ v16h load_b_frag(const _Float16* __restrict__ Bt, int ldb,
                                            int n0, int k0, int lane) {
    const int lo = lane & 15, hi = lane >> 4;
    const _Float16* p = Bt + (size_t)(n0 + lo) * ldb + k0 + hi * 16;
    return *(const v16h*)p;
}

// Same but reading f32 (softmax probabilities in LDS), converting on the fly.
__device__ __forceinline__ v16h load_b_frag_f32(const float* __restrict__ Pt, int ldb,
                                                int n0, int k0, int lane) {
    const int lo = lane & 15, hi = lane >> 4;
    const float* p = Pt + (size_t)(n0 + lo) * ldb + k0 + hi * 16;
    v16h b;
#pragma unroll
    for (int i = 0; i < 16; ++i) b[i] = (_Float16)p[i];
    return b;
}

__device__ __forceinline__ v8f wmma_f16(v16h a, v16h b, v8f c) {
    return __builtin_amdgcn_wmma_f32_16x16x32_f16(false, a, false, b, (short)0, c,
                                                  false, false);
}

// QKV epilogue: route one 16x16 D tile column (8 rows per lane) to q/k/v buffers.
__device__ __forceinline__ void qkv_store(const v8f& a, int b, int p, int obase,
                                          const float* __restrict__ bias,
                                          _Float16* __restrict__ qT,
                                          _Float16* __restrict__ kT,
                                          _Float16* __restrict__ vv) {
    const int sect = obase >> 9;              // 0=q 1=k 2=v
    const int ol = obase & (C_DIM - 1);
    const float scale = 0.04419417382415922f; // 512^-0.5 folded into q
    if (sect == 0) {
        v8h out;
#pragma unroll
        for (int r = 0; r < 8; ++r) out[r] = (_Float16)((a[r] + bias[obase + r]) * scale);
        *(v8h*)(qT + ((size_t)b * HW + p) * C_DIM + ol) = out;
    } else if (sect == 1) {
        v8h out;
#pragma unroll
        for (int r = 0; r < 8; ++r) out[r] = (_Float16)(a[r] + bias[obase + r]);
        *(v8h*)(kT + ((size_t)b * HW + p) * C_DIM + ol) = out;
    } else {
#pragma unroll
        for (int r = 0; r < 8; ++r)
            vv[((size_t)b * C_DIM + ol + r) * HW + p] = (_Float16)(a[r] + bias[obase + r]);
    }
}

// ---------------- Kernel 1: convert weights fp32 -> f16 -----------------------
__global__ void __launch_bounds__(256)
convert_w_kernel(const float* __restrict__ qkv_w, const float* __restrict__ out_w,
                 _Float16* __restrict__ qkv_h, _Float16* __restrict__ out_h) {
    int i = blockIdx.x * 256 + threadIdx.x;
    if (i < 3 * C_DIM * C_DIM) qkv_h[i] = (_Float16)qkv_w[i];
    if (i < C_DIM * C_DIM) out_h[i] = (_Float16)out_w[i];
}

// ---------------- Kernel 2: GroupNorm -> xn f16 [b][pixel][channel] -----------
__global__ void __launch_bounds__(256)
groupnorm_kernel(const float* __restrict__ x, const float* __restrict__ gamma,
                 const float* __restrict__ beta, _Float16* __restrict__ xn) {
    const int b = blockIdx.y;
    const int g = blockIdx.x;
    const int tid = threadIdx.x;
    const int CG = C_DIM / GROUPS;         // 64 channels per group
    const float* xg = x + ((size_t)b * C_DIM + g * CG) * HW;

    float s = 0.f, ss = 0.f;
    for (int e = tid; e < CG * HW; e += 256) {
        float v = xg[e];
        s += v; ss += v * v;
    }
    __shared__ float rs[256], rss[256];
    rs[tid] = s; rss[tid] = ss;
    __syncthreads();
    for (int k = 128; k > 0; k >>= 1) {
        if (tid < k) { rs[tid] += rs[tid + k]; rss[tid] += rss[tid + k]; }
        __syncthreads();
    }
    const float inv_n = 1.f / (float)(CG * HW);
    float mean = rs[0] * inv_n;
    float var  = rss[0] * inv_n - mean * mean;
    float rinv = rsqrtf(var + 1e-5f);

    for (int e = tid; e < CG * HW; e += 256) {
        int cl = e >> 10, p = e & (HW - 1);
        int c = g * CG + cl;
        float v = (xg[e] - mean) * rinv * gamma[c] + beta[c];
        xn[((size_t)b * HW + p) * C_DIM + c] = (_Float16)v;
    }
}

// ---------------- Kernel 3: QKV GEMM, wave computes 32(M) x 64(N) -------------
// D[o][p] = W[o][:] . xn[:, p]; writes Q,K pixel-major [hw][c] f16 (q pre-scaled),
// V channel-major [c][hw]. 2x4 register blocking: 8 WMMA per 6 fragment loads.
__global__ void __launch_bounds__(256)
qkv_gemm_kernel(const _Float16* __restrict__ W, const float* __restrict__ bias,
                const _Float16* __restrict__ xn,
                _Float16* __restrict__ qT, _Float16* __restrict__ kT,
                _Float16* __restrict__ vv) {
    const int lane = threadIdx.x & 31;
    const int wave = threadIdx.x >> 5;
    const int b = blockIdx.y;
    const int id = blockIdx.x * 8 + wave;     // 48*16 = 768 wave tasks / batch
    const int mt0 = (id >> 4) * 2;            // M tile pair: 0,2,..,94
    const int ng = id & 15;                   // pixel tile group of 4
    const _Float16* Bt = xn + (size_t)b * HW * C_DIM;   // [pixel][channel]

    __builtin_prefetch(W + (size_t)mt0 * 16 * C_DIM, 0, 3);

    v8f acc0[4] = {}, acc1[4] = {};
#pragma unroll 2
    for (int k0 = 0; k0 < C_DIM; k0 += 32) {
        v16h a0 = load_a_frag(W, C_DIM, mt0 * 16, k0, lane);
        v16h a1 = load_a_frag(W, C_DIM, (mt0 + 1) * 16, k0, lane);
        v16h bb[4];
#pragma unroll
        for (int j = 0; j < 4; ++j)
            bb[j] = load_b_frag(Bt, C_DIM, (ng * 4 + j) * 16, k0, lane);
#pragma unroll
        for (int j = 0; j < 4; ++j) acc0[j] = wmma_f16(a0, bb[j], acc0[j]);
#pragma unroll
        for (int j = 0; j < 4; ++j) acc1[j] = wmma_f16(a1, bb[j], acc1[j]);
    }
    const int lo = lane & 15, hi = lane >> 4;
    const int ob0 = mt0 * 16 + hi * 8;
#pragma unroll
    for (int j = 0; j < 4; ++j) {
        const int p = (ng * 4 + j) * 16 + lo;
        qkv_store(acc0[j], b, p, ob0, bias, qT, kT, vv);
        qkv_store(acc1[j], b, p, ob0 + 16, bias, qT, kT, vv);
    }
}

// ---------------- Kernel 4: attention (per batch x 32-row block) --------------
// Pass1: S = Q_blk K^T in LDS (f32). Pass2: softmax. Pass3: O = V P^T.
__global__ void __launch_bounds__(256)
attn_kernel(const _Float16* __restrict__ qT, const _Float16* __restrict__ kT,
            const _Float16* __restrict__ vv, _Float16* __restrict__ O) {
    extern __shared__ float smem[];
    float* S   = smem;                  // AROWS * HW f32 = 128 KB
    float* red = smem + AROWS * HW;     // AROWS * 8 f32

    const int b  = blockIdx.y;
    const int i0 = blockIdx.x * AROWS;
    const int lane = threadIdx.x & 31;
    const int wave = threadIdx.x >> 5;  // 8 waves

    const _Float16* Q = qT + (size_t)b * HW * C_DIM;  // [i][c] (scaled)
    const _Float16* K = kT + (size_t)b * HW * C_DIM;  // [j][c]
    const _Float16* V = vv + (size_t)b * C_DIM * HW;  // [c][j]

    __builtin_prefetch(Q + (size_t)i0 * C_DIM, 0, 3);

    // ---- Pass 1: 16 col-tile-group tasks, each computes both row tiles (2x4)
    for (int t = wave; t < 16; t += 8) {
        v8f acc0[4] = {}, acc1[4] = {};
#pragma unroll 2
        for (int k0 = 0; k0 < C_DIM; k0 += 32) {
            v16h a0 = load_a_frag(Q, C_DIM, i0, k0, lane);
            v16h a1 = load_a_frag(Q, C_DIM, i0 + 16, k0, lane);
            v16h bb[4];
#pragma unroll
            for (int j = 0; j < 4; ++j)
                bb[j] = load_b_frag(K, C_DIM, (t * 4 + j) * 16, k0, lane);
#pragma unroll
            for (int j = 0; j < 4; ++j) acc0[j] = wmma_f16(a0, bb[j], acc0[j]);
#pragma unroll
            for (int j = 0; j < 4; ++j) acc1[j] = wmma_f16(a1, bb[j], acc1[j]);
        }
        const int lo = lane & 15, hi = lane >> 4;
#pragma unroll
        for (int j = 0; j < 4; ++j) {
            const int col = (t * 4 + j) * 16 + lo;
#pragma unroll
            for (int r = 0; r < 8; ++r) {
                S[(size_t)(hi * 8 + r) * HW + col]      = acc0[j][r];
                S[(size_t)(16 + hi * 8 + r) * HW + col] = acc1[j][r];
            }
        }
    }
    __syncthreads();

    // ---- Pass 2: softmax, 8 threads per row, 128 columns each
    {
        const int row = threadIdx.x >> 3;
        const int seg = threadIdx.x & 7;
        float* Sr = S + (size_t)row * HW + seg * 128;
        float m = -1e30f;
        for (int j = 0; j < 128; ++j) m = fmaxf(m, Sr[j]);
        red[row * 8 + seg] = m;
        __syncthreads();
        float rm = red[row * 8];
#pragma unroll
        for (int k = 1; k < 8; ++k) rm = fmaxf(rm, red[row * 8 + k]);
        float s = 0.f;
        for (int j = 0; j < 128; ++j) { float e = __expf(Sr[j] - rm); Sr[j] = e; s += e; }
        __syncthreads();                 // everyone done reading maxes
        red[row * 8 + seg] = s;
        __syncthreads();
        float rsum = 0.f;
#pragma unroll
        for (int k = 0; k < 8; ++k) rsum += red[row * 8 + k];
        float inv = 1.f / rsum;
        for (int j = 0; j < 128; ++j) Sr[j] *= inv;
    }
    __syncthreads();

    // ---- Pass 3: O = V P^T : 8 tasks (one per wave); each computes 4 c-tiles x
    //      both i-tiles. P fragments (f32->f16 LDS convert) each reused 4x.
    for (int t = wave; t < 8; t += 8) {
        v8f acc0[4] = {}, acc1[4] = {};
        for (int k0 = 0; k0 < HW; k0 += 32) {
            v16h b0 = load_b_frag_f32(S, HW, 0, k0, lane);    // P rows 0..15
            v16h b1 = load_b_frag_f32(S, HW, 16, k0, lane);   // P rows 16..31
            v16h aa[4];
#pragma unroll
            for (int j = 0; j < 4; ++j)
                aa[j] = load_a_frag(V, HW, (t * 4 + j) * 16, k0, lane); // A[c][j]
#pragma unroll
            for (int j = 0; j < 4; ++j) acc0[j] = wmma_f16(aa[j], b0, acc0[j]);
#pragma unroll
            for (int j = 0; j < 4; ++j) acc1[j] = wmma_f16(aa[j], b1, acc1[j]);
        }
        const int lo = lane & 15, hi = lane >> 4;
        const int p0 = i0 + lo;
        const int p1 = i0 + 16 + lo;
#pragma unroll
        for (int j = 0; j < 4; ++j) {
            const int cbase = (t * 4 + j) * 16 + hi * 8;
            v8h out0, out1;
#pragma unroll
            for (int r = 0; r < 8; ++r) { out0[r] = (_Float16)acc0[j][r];
                                          out1[r] = (_Float16)acc1[j][r]; }
            *(v8h*)(O + ((size_t)b * HW + p0) * C_DIM + cbase) = out0;
            *(v8h*)(O + ((size_t)b * HW + p1) * C_DIM + cbase) = out1;
        }
    }
}

// ---------------- Kernel 5: out projection + bias + residual ------------------
// Wave computes 32(M) x 64(N), 2x4 register blocking.
__global__ void __launch_bounds__(256)
proj_kernel(const _Float16* __restrict__ Wh, const float* __restrict__ bias,
            const _Float16* __restrict__ O, const float* __restrict__ x,
            float* __restrict__ y) {
    const int lane = threadIdx.x & 31;
    const int wave = threadIdx.x >> 5;
    const int b = blockIdx.y;
    const int id = blockIdx.x * 8 + wave;     // 16*16 = 256 wave tasks / batch
    const int mt0 = (id >> 4) * 2;            // M tile pair: 0,2,..,30
    const int ng = id & 15;                   // pixel tile group of 4
    const _Float16* Bt = O + (size_t)b * HW * C_DIM;   // [pixel][channel]

    __builtin_prefetch(Wh + (size_t)mt0 * 16 * C_DIM, 0, 3);

    v8f acc0[4] = {}, acc1[4] = {};
#pragma unroll 2
    for (int k0 = 0; k0 < C_DIM; k0 += 32) {
        v16h a0 = load_a_frag(Wh, C_DIM, mt0 * 16, k0, lane);
        v16h a1 = load_a_frag(Wh, C_DIM, (mt0 + 1) * 16, k0, lane);
        v16h bb[4];
#pragma unroll
        for (int j = 0; j < 4; ++j)
            bb[j] = load_b_frag(Bt, C_DIM, (ng * 4 + j) * 16, k0, lane);
#pragma unroll
        for (int j = 0; j < 4; ++j) acc0[j] = wmma_f16(a0, bb[j], acc0[j]);
#pragma unroll
        for (int j = 0; j < 4; ++j) acc1[j] = wmma_f16(a1, bb[j], acc1[j]);
    }
    const int lo = lane & 15, hi = lane >> 4;
    const int ob0 = mt0 * 16 + hi * 8;
#pragma unroll
    for (int j = 0; j < 4; ++j) {
        const int p = (ng * 4 + j) * 16 + lo;
#pragma unroll
        for (int r = 0; r < 8; ++r) {
            size_t i0x = ((size_t)b * C_DIM + ob0 + r) * HW + p;
            size_t i1x = ((size_t)b * C_DIM + ob0 + 16 + r) * HW + p;
            y[i0x] = x[i0x] + acc0[j][r] + bias[ob0 + r];
            y[i1x] = x[i1x] + acc1[j][r] + bias[ob0 + 16 + r];
        }
    }
}

// ---------------- Launch ------------------------------------------------------
extern "C" void kernel_launch(void* const* d_in, const int* in_sizes, int n_in,
                              void* d_out, int out_size, void* d_ws, size_t ws_size,
                              hipStream_t stream) {
    const float* x          = (const float*)d_in[0];
    const float* norm_gamma = (const float*)d_in[1];
    const float* norm_beta  = (const float*)d_in[2];
    const float* qkv_w      = (const float*)d_in[3];
    const float* qkv_b      = (const float*)d_in[4];
    const float* out_w      = (const float*)d_in[5];
    const float* out_b      = (const float*)d_in[6];
    float* y = (float*)d_out;

    // workspace layout (f16 elements)
    const size_t NTENS = (size_t)B_DIM * HW * C_DIM;   // 16,777,216
    _Float16* ws  = (_Float16*)d_ws;
    _Float16* xn  = ws;
    _Float16* qT  = xn + NTENS;
    _Float16* kT  = qT + NTENS;
    _Float16* vv  = kT + NTENS;
    _Float16* O   = vv + NTENS;
    _Float16* qkv_h = O + NTENS;
    _Float16* out_h = qkv_h + (size_t)3 * C_DIM * C_DIM;

    // 1) weight conversion
    convert_w_kernel<<<dim3((3 * C_DIM * C_DIM + 255) / 256), dim3(256), 0, stream>>>(
        qkv_w, out_w, qkv_h, out_h);

    // 2) groupnorm -> xn [b][p][c] f16
    groupnorm_kernel<<<dim3(GROUPS, B_DIM), dim3(256), 0, stream>>>(
        x, norm_gamma, norm_beta, xn);

    // 3) qkv gemm -> qT/kT [b][p][c], v [b][c][p]
    qkv_gemm_kernel<<<dim3(48 * 16 / 8, B_DIM), dim3(256), 0, stream>>>(
        qkv_h, qkv_b, xn, qT, kT, vv);

    // 4) attention: 32 rows per block, 128KB + 1KB LDS
    const size_t attn_lds = (size_t)AROWS * HW * sizeof(float) + AROWS * 8 * sizeof(float);
    hipFuncSetAttribute((const void*)attn_kernel,
                        hipFuncAttributeMaxDynamicSharedMemorySize, (int)attn_lds);
    attn_kernel<<<dim3(HW / AROWS, B_DIM), dim3(256), attn_lds, stream>>>(qT, kT, vv, O);

    // 5) projection + residual
    proj_kernel<<<dim3(16 * 16 / 8, B_DIM), dim3(256), 0, stream>>>(
        out_h, out_b, O, x, y);
}
